// CustomRNN_20186346291415
// MI455X (gfx1250) — matrix operand order repaired and checked by reading, not compiled, not run on hardware
//
#include <hip/hip_runtime.h>
#include <hip/hip_bf16.h>

// ---------------------------------------------------------------------------
// 2-layer tanh RNN on gfx1250 using v_wmma_f32_16x16x32_bf16.
// B=64, S=512, D=H=1024.
// ---------------------------------------------------------------------------

#define RNN_B 64
#define RNN_S 512
#define RNN_D 1024
#define RNN_H 1024

typedef __attribute__((ext_vector_type(16))) __bf16 bf16x16;
typedef __attribute__((ext_vector_type(8)))  __bf16 bf16x8;
typedef __attribute__((ext_vector_type(8)))  float  floatx8;

__device__ __forceinline__ __bf16 f2bf(float f) {
    union { float f; unsigned u; } v; v.f = f;
    unsigned r = v.u + 0x7FFFu + ((v.u >> 16) & 1u);   // round-to-nearest-even
    unsigned short h = (unsigned short)(r >> 16);
    __bf16 b;
    __builtin_memcpy(&b, &h, 2);
    return b;
}

// Branchless tanh: 1 - 2/(exp2(2*log2e*x)+1). Saturates to +/-1 for large |x|
// (rcp(inf)=0), NaN propagates. One v_exp_f32 + one v_rcp_f32, no EXEC forks.
__device__ __forceinline__ float tanh_fast(float x) {
    float e = __builtin_amdgcn_exp2f(x * 2.8853900817779268f);
    float r = __builtin_amdgcn_rcpf(e + 1.0f);
    return __builtin_fmaf(-2.0f, r, 1.0f);
}

// ---------------------------------------------------------------------------
// Prep kernels
// ---------------------------------------------------------------------------

__global__ void cvt_weights(const float* __restrict__ w0, const float* __restrict__ w1,
                            const float* __restrict__ w2, const float* __restrict__ w3,
                            __bf16* __restrict__ o0, __bf16* __restrict__ o1,
                            __bf16* __restrict__ o2, __bf16* __restrict__ o3, int n) {
    int i = blockIdx.x * blockDim.x + threadIdx.x;
    if (i < n) {
        o0[i] = f2bf(w0[i]);
        o1[i] = f2bf(w1[i]);
        o2[i] = f2bf(w2[i]);
        o3[i] = f2bf(w3[i]);
    }
}

__global__ void cvt_x(const float* __restrict__ x, __bf16* __restrict__ xb, long n) {
    long i = (long)blockIdx.x * blockDim.x + threadIdx.x;
    if (i < n) xb[i] = f2bf(x[i]);
}

__global__ void make_bias(const float* __restrict__ bi0, const float* __restrict__ bh0,
                          const float* __restrict__ bi1, const float* __restrict__ bh1,
                          float* __restrict__ bias0, float* __restrict__ bias1, int n) {
    int i = blockIdx.x * blockDim.x + threadIdx.x;
    if (i < n) {
        bias0[i] = bi0[i] + bh0[i];
        bias1[i] = bi1[i] + bh1[i];
    }
}

__global__ void zero_h(__bf16* __restrict__ h0, __bf16* __restrict__ h1, int n) {
    int i = blockIdx.x * blockDim.x + threadIdx.x;
    if (i < n) {
        unsigned short z = 0;
        __builtin_memcpy(&h0[i], &z, 2);
        __builtin_memcpy(&h1[i], &z, 2);
    }
}

// ---------------------------------------------------------------------------
// Fused RNN cell:  Hnew = tanh( A * Wi^T  +  Hin * Wh^T  +  bias )
//   A    : [64 x *] bf16 (fast path) or fp32 (fallback, inline convert)
//   Wi/Wh: [1024 x 1024] bf16 row-major (row = output neuron) -> B operand
//          columns are W rows => two 16-byte lane loads (k and k+16).
//   Hin  : [64 x 1024] bf16
//   Out  : fp32 (Fout, row stride strideF) + bf16 (Hout, next-step operand)
//
// One wave = one 16x16 output tile; 64 blocks x 128 threads = 256 waves
// (4 M-tiles x 64 N-tiles). K loop: 32 iterations of two
// v_wmma_f32_16x16x32_bf16 (ih + hh) sharing one f32 accumulator.
// ---------------------------------------------------------------------------

template <bool A_IS_F32>
__global__ void __launch_bounds__(128)
rnn_cell(const void* __restrict__ A, long strideA,
         const __bf16* __restrict__ Wi, const __bf16* __restrict__ Wh,
         const __bf16* __restrict__ Hin, const float* __restrict__ bias,
         __bf16* __restrict__ Hout, float* __restrict__ Fout, long strideF) {
    const int lane = threadIdx.x & 31;
    const int wave = threadIdx.x >> 5;
    const int gw   = blockIdx.x * 4 + wave;      // 0..255
    const int Mbase = (gw & 3) << 4;             // 0,16,32,48
    const int Nbase = (gw >> 2) << 4;            // 0..1008

    const int nl   = lane & 15;
    const int half = lane >> 4;
    const int kbA  = half * 8;                   // A-frag K sub-offset (+16 for hi half)
    const int kbB  = half * 16;                  // unused; B frags load at kbA like A
    const int mA   = Mbase + nl;                 // A/H row held by this lane
    const int nB   = Nbase + nl;                 // W row (= output column)

    const __bf16* Hrow  = Hin + (long)mA * RNN_H;
    const __bf16* WiRow = Wi  + (long)nB * RNN_H;
    const __bf16* WhRow = Wh  + (long)nB * RNN_H;

    floatx8 acc = {};

    for (int k0 = 0; k0 < RNN_H; k0 += 32) {
        // ---- A (input) fragment: 16x32 bf16 ----
        union { bf16x16 v; bf16x8 h[2]; } a_in;
        if (A_IS_F32) {
            const float* Ar = (const float*)A + (long)mA * strideA + k0;
            #pragma unroll
            for (int i = 0; i < 8; ++i) a_in.v[i]     = f2bf(Ar[kbA + i]);
            #pragma unroll
            for (int i = 0; i < 8; ++i) a_in.v[8 + i] = f2bf(Ar[kbA + 16 + i]);
        } else {
            const __bf16* Ar = (const __bf16*)A + (long)mA * strideA + k0;
            a_in.h[0] = *(const bf16x8*)(Ar + kbA);
            a_in.h[1] = *(const bf16x8*)(Ar + kbA + 16);
        }

        // ---- H (recurrent) fragment: 16x32 bf16 ----
        union { bf16x16 v; bf16x8 h[2]; } a_h;
        a_h.h[0] = *(const bf16x8*)(Hrow + k0 + kbA);
        a_h.h[1] = *(const bf16x8*)(Hrow + k0 + kbA + 16);

        // ---- B fragments: 32x16 bf16, two 16B loads per lane (same K order as A) ----
        union { bf16x16 v; bf16x8 h[2]; } b_wi, b_wh;
        b_wi.h[0] = *(const bf16x8*)(WiRow + k0 + kbA);
        b_wi.h[1] = *(const bf16x8*)(WiRow + k0 + kbA + 16);
        b_wh.h[0] = *(const bf16x8*)(WhRow + k0 + kbA);
        b_wh.h[1] = *(const bf16x8*)(WhRow + k0 + kbA + 16);

        acc = __builtin_amdgcn_wmma_f32_16x16x32_bf16(
                  false, a_in.v, false, b_wi.v, (short)0, acc, false, false);
        acc = __builtin_amdgcn_wmma_f32_16x16x32_bf16(
                  false, a_h.v,  false, b_wh.v, (short)0, acc, false, false);
    }

    // ---- epilogue: bias + tanh, write fp32 + bf16 ----
    const float bv = bias[nB];
    #pragma unroll
    for (int r = 0; r < 8; ++r) {
        const int m = Mbase + half * 8 + r;       // C/D layout: lanes 16-31 -> M+8
        const float v = tanh_fast(acc[r] + bv);
        Fout[(long)m * strideF + nB] = v;
        Hout[(long)m * RNN_H + nB]   = f2bf(v);
    }
}

// ---------------------------------------------------------------------------
// Final: hT[0] = h0_final (fp32 buffer), hT[1] = outs[:, S-1, :]
// ---------------------------------------------------------------------------
__global__ void write_hT(const float* __restrict__ h0f, const float* __restrict__ outs,
                         float* __restrict__ tail, int n) {
    int i = blockIdx.x * blockDim.x + threadIdx.x;
    if (i < n) {
        int b = i / RNN_H, c = i % RNN_H;
        tail[i]     = h0f[i];
        tail[n + i] = outs[((long)b * RNN_S + (RNN_S - 1)) * RNN_H + c];
    }
}

// ---------------------------------------------------------------------------
// Launcher
// ---------------------------------------------------------------------------
extern "C" void kernel_launch(void* const* d_in, const int* in_sizes, int n_in,
                              void* d_out, int out_size, void* d_ws, size_t ws_size,
                              hipStream_t stream) {
    (void)in_sizes; (void)n_in; (void)out_size;

    const float* x     = (const float*)d_in[0];
    const float* W_ih0 = (const float*)d_in[1];
    const float* b_ih0 = (const float*)d_in[2];
    const float* W_hh0 = (const float*)d_in[3];
    const float* b_hh0 = (const float*)d_in[4];
    const float* W_ih1 = (const float*)d_in[5];
    const float* b_ih1 = (const float*)d_in[6];
    const float* W_hh1 = (const float*)d_in[7];
    const float* b_hh1 = (const float*)d_in[8];

    float* out = (float*)d_out;                       // [B,S,H] outs, then hT[2,B,H]

    // ---- workspace carve-up (256B aligned slabs) ----
    char* ws = (char*)d_ws;
    size_t off = 0;
    auto alloc = [&](size_t bytes) -> char* {
        char* p = ws + off;
        off += (bytes + 255) & ~(size_t)255;
        return p;
    };
    const size_t WBYTES = (size_t)RNN_H * RNN_H * sizeof(__bf16);   // 2 MB
    __bf16* Wb_ih0 = (__bf16*)alloc(WBYTES);
    __bf16* Wb_hh0 = (__bf16*)alloc(WBYTES);
    __bf16* Wb_ih1 = (__bf16*)alloc(WBYTES);
    __bf16* Wb_hh1 = (__bf16*)alloc(WBYTES);
    float*  bias0  = (float*)alloc(RNN_H * sizeof(float));
    float*  bias1  = (float*)alloc(RNN_H * sizeof(float));
    const size_t HBYTES = (size_t)RNN_B * RNN_H * sizeof(__bf16);   // 128 KB
    __bf16* h0b[2] = { (__bf16*)alloc(HBYTES), (__bf16*)alloc(HBYTES) };
    __bf16* h1b[2] = { (__bf16*)alloc(HBYTES), (__bf16*)alloc(HBYTES) };
    float*  h0f    = (float*)alloc((size_t)RNN_B * RNN_H * sizeof(float));

    // Optional slab: whole-x bf16 copy (64 MB). Use it if ws is big enough;
    // deterministic across calls (ws_size is fixed).
    const long XN = (long)RNN_B * RNN_S * RNN_D;
    __bf16* xb = (__bf16*)(ws + off);
    const bool pre_cvt_x = (off + (size_t)XN * sizeof(__bf16)) <= ws_size;

    // ---- one-time prep ----
    {
        const int n = RNN_H * RNN_H;
        cvt_weights<<<(n + 255) / 256, 256, 0, stream>>>(
            W_ih0, W_hh0, W_ih1, W_hh1, Wb_ih0, Wb_hh0, Wb_ih1, Wb_hh1, n);
        make_bias<<<(RNN_H + 255) / 256, 256, 0, stream>>>(
            b_ih0, b_hh0, b_ih1, b_hh1, bias0, bias1, RNN_H);
        const int hn = RNN_B * RNN_H;
        zero_h<<<(hn + 255) / 256, 256, 0, stream>>>(h0b[0], h1b[0], hn);
        if (pre_cvt_x) {
            cvt_x<<<(unsigned)((XN + 255) / 256), 256, 0, stream>>>(x, xb, XN);
        }
    }

    // ---- sequential scan over time ----
    dim3 grid(64), block(128);                        // 256 waves = 4 Mtiles x 64 Ntiles
    for (int t = 0; t < RNN_S; ++t) {
        const int cur = t & 1, nxt = (t + 1) & 1;
        // layer 0: h0 = tanh(x_t @ Wih0^T + h0 @ Whh0^T + bias0)
        if (pre_cvt_x) {
            rnn_cell<false><<<grid, block, 0, stream>>>(
                (const void*)(xb + (long)t * RNN_D), (long)RNN_S * RNN_D,
                Wb_ih0, Wb_hh0, h0b[cur], bias0,
                h0b[nxt], h0f, (long)RNN_H);
        } else {
            rnn_cell<true><<<grid, block, 0, stream>>>(
                (const void*)(x + (long)t * RNN_D), (long)RNN_S * RNN_D,
                Wb_ih0, Wb_hh0, h0b[cur], bias0,
                h0b[nxt], h0f, (long)RNN_H);
        }
        // layer 1: h1 = tanh(h0 @ Wih1^T + h1 @ Whh1^T + bias1); fp32 -> outs[:,t,:]
        rnn_cell<false><<<grid, block, 0, stream>>>(
            (const void*)h0b[nxt], (long)RNN_H,
            Wb_ih1, Wb_hh1, h1b[cur], bias1,
            h1b[nxt], out + (long)t * RNN_H, (long)RNN_S * RNN_H);
    }

    // ---- hT tail ----
    {
        const int n = RNN_B * RNN_H;
        write_hT<<<(n + 255) / 256, 256, 0, stream>>>(
            h0f, out, out + (long)RNN_B * RNN_S * RNN_H, n);
    }
}
